// LSTM_3968549781697
// MI455X (gfx1250) — matrix operand order repaired and checked by reading, not compiled, not run on hardware
//
#include <hip/hip_runtime.h>

typedef __attribute__((ext_vector_type(16))) _Float16     v16h;
typedef __attribute__((ext_vector_type(8)))  float        v8f;
typedef __attribute__((ext_vector_type(8)))  unsigned int v8u;

namespace {
constexpr int kB = 4096;
constexpr int kT = 512;
constexpr int kI = 8;
constexpr int kH = 10;

// Branch-free activations on gfx1250's V_TANH_F32 (exp2/rcp fallback).
__device__ __forceinline__ float fast_tanh(float x) {
#if __has_builtin(__builtin_amdgcn_tanhf)
  return __builtin_amdgcn_tanhf(x);
#else
  const float e = __builtin_amdgcn_exp2f(x * 2.88539008f);  // 2*log2(e)
  return 1.0f - 2.0f * __builtin_amdgcn_rcpf(e + 1.0f);
#endif
}
// sigmoid(g + b) with half-bias hb = 0.5*b folded into the scaling fma:
// sig(g+b) = 0.5*tanh(0.5*g + hb) + 0.5
__device__ __forceinline__ float fast_sig_b(float g, float hb) {
#if __has_builtin(__builtin_amdgcn_tanhf)
  return __builtin_fmaf(0.5f, __builtin_amdgcn_tanhf(__builtin_fmaf(0.5f, g, hb)), 0.5f);
#else
  const float x = __builtin_fmaf(0.5f, g, hb) * 2.0f;
  return __builtin_amdgcn_rcpf(1.0f + __builtin_amdgcn_exp2f(x * -1.44269504f));
#endif
}
}  // namespace

// One wave (32 threads) per block; each wave owns 16 batch rows and runs the
// entire T=512 two-layer recurrence fused, with gates computed by
// v_wmma_f32_16x16x32_f16 (one 16x16 N-tile per gate i/f/g/o, C = inline 0,
// bias folded into the activation stage).
//
// Feature (A-operand) layouts, K padded to 32 in 16-wide groups so that every
// lane's h-store is unconditional (unit>=10 lanes produce exactly h=0 and
// write into zero padding):
//   featA0: [ x(0..7) | pad(8..15) | h0(16..25) | pad(26..31) ]
//   featA1: [ h1(0..9) | pad(10..15) | h2(16..25) | pad(26..31) ]
__global__ __launch_bounds__(32)
void lstm2_fused_wmma(const float* __restrict__ x,
                      const float* __restrict__ W_ih0, const float* __restrict__ W_hh0,
                      const float* __restrict__ b_ih0, const float* __restrict__ b_hh0,
                      const float* __restrict__ W_ih1, const float* __restrict__ W_hh1,
                      const float* __restrict__ b_ih1, const float* __restrict__ b_hh1,
                      const float* __restrict__ fc_w,  const float* __restrict__ fc_b,
                      float* __restrict__ out)
{
  __shared__ _Float16 W0lds[64 * 32];   // 64 gate-cols x 32 features
  __shared__ _Float16 W1lds[64 * 32];
  __shared__ float    bias0lds[64];
  __shared__ float    bias1lds[64];
  __shared__ _Float16 featA0[16 * 32];  // per-wave A images (16 rows x 32 K)
  __shared__ _Float16 featA1[16 * 32];
  __shared__ float    red[32][8];

  const int lane   = threadIdx.x;   // 0..31
  const int col    = lane & 15;     // N column within a 16x16 tile (= unit j)
  const int halfid = lane >> 4;     // lane half selects K sub-range
  const int b0     = blockIdx.x * 16;

  // ---- build f16 weight images + fused biases in LDS ----
  for (int idx = lane; idx < 64 * 32; idx += 32) {
    const int n = idx >> 5;  // padded gate-column 0..63
    const int k = idx & 31;  // feature 0..31
    const int q = n >> 4;    // gate: 0=i 1=f 2=g 3=o (PyTorch order)
    const int j = n & 15;    // unit within gate
    float w0 = 0.0f, w1 = 0.0f;
    if (j < kH) {
      const int row = q * kH + j;
      if (k < kI)                      w0 = W_ih0[row * kI + k];
      else if (k >= 16 && k < 16 + kH) w0 = W_hh0[row * kH + (k - 16)];
      if (k < kH)                      w1 = W_ih1[row * kH + k];
      else if (k >= 16 && k < 16 + kH) w1 = W_hh1[row * kH + (k - 16)];
    }
    W0lds[idx] = (_Float16)w0;
    W1lds[idx] = (_Float16)w1;
  }
  for (int n = lane; n < 64; n += 32) {
    const int q = n >> 4, j = n & 15;
    float v0 = 0.0f, v1 = 0.0f;
    if (j < kH) {
      v0 = b_ih0[q * kH + j] + b_hh0[q * kH + j];
      v1 = b_ih1[q * kH + j] + b_hh1[q * kH + j];
    }
    bias0lds[n] = v0; bias1lds[n] = v1;
  }
  for (int idx = lane; idx < 16 * 32; idx += 32) {
    featA0[idx] = (_Float16)0.0f;   // zero h state and all padding
    featA1[idx] = (_Float16)0.0f;
  }
  __syncthreads();

  // ---- per-lane constant B fragments (4 gate tiles per layer) ----
  // 16-bit B 32x16 layout: lane = column n; VGPR0-3: K = halfid*8 + 2r..,
  // VGPR4-7: K = 16 + halfid*8 + 2(r-4).., two K per dword.
  const unsigned int* W0u = reinterpret_cast<const unsigned int*>(W0lds);
  const unsigned int* W1u = reinterpret_cast<const unsigned int*>(W1lds);
  v16h bf0[4], bf1[4];
#pragma unroll
  for (int q = 0; q < 4; ++q) {
    v8u t0, t1;
#pragma unroll
    for (int r = 0; r < 8; ++r) {
      const int uidx = (q * 16 + col) * 16 + (r >> 2) * 8 + halfid * 4 + (r & 3);  // row stride: 16 dwords
      t0[r] = W0u[uidx];
      t1[r] = W1u[uidx];
    }
    bf0[q] = __builtin_bit_cast(v16h, t0);
    bf1[q] = __builtin_bit_cast(v16h, t1);
  }
  // Per-lane bias scalars (half-biases for sigmoid gates, full for tanh gate).
  const float hb0_i = 0.5f * bias0lds[ 0 + col];
  const float hb0_f = 0.5f * bias0lds[16 + col];
  const float  b0_g =        bias0lds[32 + col];
  const float hb0_o = 0.5f * bias0lds[48 + col];
  const float hb1_i = 0.5f * bias1lds[ 0 + col];
  const float hb1_f = 0.5f * bias1lds[16 + col];
  const float  b1_g =        bias1lds[32 + col];
  const float hb1_o = 0.5f * bias1lds[48 + col];
  const float fcwj = (col < kH) ? fc_w[col] : 0.0f;

  // Per-lane state: C/D tile layout puts rows m = r + 8*halfid, column = col.
  float c0[8], c1[8], acc[8];
#pragma unroll
  for (int r = 0; r < 8; ++r) { c0[r] = 0.0f; c1[r] = 0.0f; acc[r] = 0.0f; }

  const unsigned int* fa0u = reinterpret_cast<const unsigned int*>(featA0);
  const unsigned int* fa1u = reinterpret_cast<const unsigned int*>(featA1);
  const int srow = lane >> 1;      // x staging: 2 lanes per batch row
  const int part = lane & 1;       // which float4 of the 8 inputs
  const float4* x4 = reinterpret_cast<const float4*>(x);
  const v8f zc = {};               // zero C operand (inline-0 SRC2)

  for (int t = 0; t < kT; ++t) {
    // stage x_t (16 rows x 8 f32 -> f16) into featA0 cols 0..7
    const size_t xi = ((size_t)(b0 + srow) * kT + t) * 2 + part;
    const float4 xv = x4[xi];
    __builtin_prefetch(&x4[xi + 2], 0, 3);  // next timestep's row, keep near
    {
      const int base = srow * 32 + part * 4;
      featA0[base + 0] = (_Float16)xv.x;
      featA0[base + 1] = (_Float16)xv.y;
      featA0[base + 2] = (_Float16)xv.z;
      featA0[base + 3] = (_Float16)xv.w;
    }
    __syncthreads();  // single-wave WG: hw NOP, compiler LDS fence

    // ---- layer 0: A = [x_t | h0_prev] ----
    // 16-bit A 16x32 layout: row = lane%16; VGPR0-3: K = halfid*8 + 2r..,
    // VGPR4-7: K = 16 + halfid*8 + 2(r-4)..
    v8u ta;
#pragma unroll
    for (int r = 0; r < 8; ++r) {
      const int uidx = col * 16 + (r >> 2) * 8 + halfid * 4 + (r & 3);
      ta[r] = fa0u[uidx];
    }
    const v16h a0 = __builtin_bit_cast(v16h, ta);

    v8f g0, g1, g2, g3;
    g0 = __builtin_amdgcn_wmma_f32_16x16x32_f16(false, a0, false, bf0[0], (short)0, zc, false, false);
    g1 = __builtin_amdgcn_wmma_f32_16x16x32_f16(false, a0, false, bf0[1], (short)0, zc, false, false);
    g2 = __builtin_amdgcn_wmma_f32_16x16x32_f16(false, a0, false, bf0[2], (short)0, zc, false, false);
    g3 = __builtin_amdgcn_wmma_f32_16x16x32_f16(false, a0, false, bf0[3], (short)0, zc, false, false);

#pragma unroll
    for (int r = 0; r < 8; ++r) {
      const float iv = fast_sig_b(g0[r], hb0_i);
      const float fv = fast_sig_b(g1[r], hb0_f);
      const float gv = fast_tanh(g2[r] + b0_g);
      const float ov = fast_sig_b(g3[r], hb0_o);
      c0[r] = fv * c0[r] + iv * gv;
      const float h1 = ov * fast_tanh(c0[r]);
      const int m = r + halfid * 8;
      const _Float16 hh = (_Float16)h1;     // exactly 0 for col >= 10
      featA0[m * 32 + 16 + col] = hh;       // layer0 recurrence (K 16..25 | pad)
      featA1[m * 32 + col]      = hh;       // layer1 input      (K 0..9   | pad)
    }
    __syncthreads();

    // ---- layer 1: A = [h1_t | h2_prev] ----
    v8u tb;
#pragma unroll
    for (int r = 0; r < 8; ++r) {
      const int uidx = col * 16 + (r >> 2) * 8 + halfid * 4 + (r & 3);
      tb[r] = fa1u[uidx];
    }
    const v16h a1 = __builtin_bit_cast(v16h, tb);

    g0 = __builtin_amdgcn_wmma_f32_16x16x32_f16(false, a1, false, bf1[0], (short)0, zc, false, false);
    g1 = __builtin_amdgcn_wmma_f32_16x16x32_f16(false, a1, false, bf1[1], (short)0, zc, false, false);
    g2 = __builtin_amdgcn_wmma_f32_16x16x32_f16(false, a1, false, bf1[2], (short)0, zc, false, false);
    g3 = __builtin_amdgcn_wmma_f32_16x16x32_f16(false, a1, false, bf1[3], (short)0, zc, false, false);

#pragma unroll
    for (int r = 0; r < 8; ++r) {
      const float iv = fast_sig_b(g0[r], hb1_i);
      const float fv = fast_sig_b(g1[r], hb1_f);
      const float gv = fast_tanh(g2[r] + b1_g);
      const float ov = fast_sig_b(g3[r], hb1_o);
      c1[r] = fv * c1[r] + iv * gv;
      const float h2 = ov * fast_tanh(c1[r]);
      acc[r] += fcwj * h2;                            // fused fc_w dot over time
      featA1[(r + halfid * 8) * 32 + 16 + col] = (_Float16)h2;  // h2 recurrence
    }
    __syncthreads();
  }

  // ---- cross-lane reduction of acc over units j, then mean + bias ----
#pragma unroll
  for (int r = 0; r < 8; ++r) red[lane][r] = acc[r];
  __syncthreads();
  if (lane < 16) {
    const int grp = lane >> 3;   // rows 0-7 live in lanes 0-15, rows 8-15 in 16-31
    const int rr  = lane & 7;
    float s = 0.0f;
    for (int q = 0; q < 16; ++q) s += red[grp * 16 + q][rr];
    out[b0 + lane] = s * (1.0f / kT) + fc_b[0];
  }
}

extern "C" void kernel_launch(void* const* d_in, const int* in_sizes, int n_in,
                              void* d_out, int out_size, void* d_ws, size_t ws_size,
                              hipStream_t stream) {
  const float* x     = (const float*)d_in[0];
  const float* W_ih0 = (const float*)d_in[1];
  const float* W_hh0 = (const float*)d_in[2];
  const float* b_ih0 = (const float*)d_in[3];
  const float* b_hh0 = (const float*)d_in[4];
  const float* W_ih1 = (const float*)d_in[5];
  const float* W_hh1 = (const float*)d_in[6];
  const float* b_ih1 = (const float*)d_in[7];
  const float* b_hh1 = (const float*)d_in[8];
  const float* fc_w  = (const float*)d_in[9];
  const float* fc_b  = (const float*)d_in[10];
  float* outp = (float*)d_out;

  dim3 grid(kB / 16);   // 256 waves, one 16-row batch tile each
  dim3 block(32);       // one wave32 per block
  hipLaunchKernelGGL(lstm2_fused_wmma, grid, block, 0, stream,
                     x, W_ih0, W_hh0, b_ih0, b_hh0,
                     W_ih1, W_hh1, b_ih1, b_hh1, fc_w, fc_b, outp);
}
